// _MultiheadAttention_48704929136828
// MI455X (gfx1250) — matrix operand order repaired and checked by reading, not compiled, not run on hardware
//
#include <hip/hip_runtime.h>
#include <hip/hip_bf16.h>
#include <math.h>

// Problem constants (match reference)
#define B_  8
#define L_  1024
#define D_  1024
#define H_  16
#define DK_ 64
#define DV_ 64
#define HD_ 1024   // H_*DK_ == H_*DV_

typedef __attribute__((ext_vector_type(16))) __bf16 v16bf;
typedef __attribute__((ext_vector_type(8)))  __bf16 v8bf;
typedef __attribute__((ext_vector_type(8)))  float  v8f;

typedef __attribute__((address_space(3))) void lds_void;

static __device__ inline v8f wmma_bf16(v16bf a, v16bf b, v8f c) {
  // D = A(16x32 bf16) * B(32x16 bf16) + C(16x16 f32)
  return __builtin_amdgcn_wmma_f32_16x16x32_bf16(
      /*neg_a=*/false, a, /*neg_b=*/false, b,
      /*c_mod=*/(short)0, c, /*reuse_a=*/false, /*reuse_b=*/false);
}

// ---- CDNA5 async global->LDS copy (ASYNCcnt-tracked, bypasses VGPRs) -----
// VDST = LDS byte address (AS3 pointer value). Passing the pointer as an asm
// operand makes it escape, so the "memory" clobber keeps the LDS tile alive.
static __device__ inline void async_load_b128(__bf16* lds_dst, const void* gaddr) {
  asm volatile("global_load_async_to_lds_b128 %0, %1, off"
               :: "v"((lds_void*)lds_dst), "v"(gaddr)
               : "memory");
}
static __device__ inline void async_wait0() {
  asm volatile("s_wait_asynccnt 0x0" ::: "memory");
}

// ---- LDS fragment loaders ------------------------------------------------
// A fragment (16x32 bf16, tile stored row-major [m][k], ld elements/row).
// Lane l<16: M=l, elems 0..7 = K0..7, elems 8..15 = K16..23.
// Lane l>=16: M=l-16, K offset +8.
static __device__ inline v16bf frag_a(const __bf16* lds, int row0, int lane, int ld) {
  const __bf16* p = lds + (size_t)(row0 + (lane & 15)) * ld + ((lane & 16) ? 8 : 0);
  v8bf lo = *(const v8bf*)(p);
  v8bf hi = *(const v8bf*)(p + 16);
  v16bf r;
#pragma unroll
  for (int i = 0; i < 8; ++i) { r[i] = lo[i]; r[i + 8] = hi[i]; }
  return r;
}
// B fragment (32x16 bf16), tile stored TRANSPOSED in LDS: [n][k].
// Lane l<16: N=l, elems 0..7 = K0..7, elems 8..15 = K16..23. Lane l>=16: K offset +8.
static __device__ inline v16bf frag_b(const __bf16* lds, int col0, int lane, int ld) {
  const __bf16* p = lds + (size_t)(col0 + (lane & 15)) * ld + ((lane & 16) ? 8 : 0);
  v8bf lo = *(const v8bf*)(p);
  v8bf hi = *(const v8bf*)(p + 16);
  v16bf r;
#pragma unroll
  for (int i = 0; i < 8; ++i) { r[i] = lo[i]; r[i + 8] = hi[i]; }
  return r;
}

// ---- Global -> LDS tile loaders (block = 256 threads) --------------------
// A tile: 128 rows x 32 k, fp32 source -> bf16 LDS [m][40] (VALU convert path)
static __device__ inline void load_tile_a(const float* A, __bf16* lds, int lda, int t) {
  const int col = (t & 7) * 4;
  const int r0  = t >> 3;
#pragma unroll
  for (int i = 0; i < 4; ++i) {
    const int r = r0 + i * 32;
    float4 f = *(const float4*)(A + (size_t)r * lda + col);
    __bf16* d = lds + r * 40 + col;
    d[0] = (__bf16)f.x; d[1] = (__bf16)f.y; d[2] = (__bf16)f.z; d[3] = (__bf16)f.w;
  }
}
// A tile: 128 rows x 32 k, bf16 source -> bf16 LDS [m][40] (async DMA path)
static __device__ inline void load_tile_a(const __bf16* A, __bf16* lds, int lda, int t) {
  const int col = (t & 3) * 8;
  const int r0  = t >> 2;
#pragma unroll
  for (int i = 0; i < 2; ++i) {
    const int r = r0 + i * 64;
    async_load_b128(lds + r * 40 + col, A + (size_t)r * lda + col);
  }
}
// B tile: 32 k-rows x 128 n, fp32 source -> TRANSPOSED bf16 LDS [n][40]
static __device__ inline void load_tile_b(const float* Bm, __bf16* lds, int ldb, int t) {
  const int k  = t & 31;
  const int n0 = (t >> 5) * 16;
#pragma unroll
  for (int j = 0; j < 4; ++j) {
    float4 f = *(const float4*)(Bm + (size_t)k * ldb + n0 + j * 4);
    lds[(n0 + j * 4 + 0) * 40 + k] = (__bf16)f.x;
    lds[(n0 + j * 4 + 1) * 40 + k] = (__bf16)f.y;
    lds[(n0 + j * 4 + 2) * 40 + k] = (__bf16)f.z;
    lds[(n0 + j * 4 + 3) * 40 + k] = (__bf16)f.w;
  }
}

static __device__ inline void store_out(float* p, float v)  { *p = v; }
static __device__ inline void store_out(__bf16* p, float v) { *p = (__bf16)v; }

// ---- K1 / K5: generic C[M,N] = A[M,K] * B[K,N] ---------------------------
// 128x128 block tile, 8 wave32 in a 2(M)x4(N) grid, 64x32 per wave.
template <typename AT, typename OT>
__global__ __launch_bounds__(256)
void gemm_kernel(const AT* __restrict__ A, const float* __restrict__ Bm,
                 OT* __restrict__ C, int Kd, int lda, int ldb, int ldc) {
  __shared__ __bf16 SMEM[128 * 40 * 2];     // [Al | Bl]
  __bf16* Al = SMEM;
  __bf16* Bl = SMEM + 128 * 40;
  const int t = threadIdx.x, lane = t & 31, wid = t >> 5;
  const int wm = wid >> 2, wn = wid & 3;
  const int brow = blockIdx.y * 128, bcol = blockIdx.x * 128;
  v8f acc[4][2] = {};
  for (int k0 = 0; k0 < Kd; k0 += 32) {
    load_tile_a(A + (size_t)brow * lda + k0, Al, lda, t);
    load_tile_b(Bm + (size_t)k0 * ldb + bcol, Bl, ldb, t);
    if (k0 + 32 < Kd) {
      __builtin_prefetch(A + (size_t)brow * lda + (k0 + 32), 0, 0);
      __builtin_prefetch(Bm + (size_t)(k0 + 32) * ldb + bcol, 0, 0);
    }
    async_wait0();                           // NOP for the fp32-A instantiation
    __syncthreads();
    v16bf af[4], bfr[2];
#pragma unroll
    for (int i = 0; i < 4; ++i) af[i] = frag_a(Al, wm * 64 + i * 16, lane, 40);
#pragma unroll
    for (int j = 0; j < 2; ++j) bfr[j] = frag_b(Bl, wn * 32 + j * 16, lane, 40);
#pragma unroll
    for (int i = 0; i < 4; ++i)
#pragma unroll
      for (int j = 0; j < 2; ++j)
        acc[i][j] = wmma_bf16(af[i], bfr[j], acc[i][j]);
    __syncthreads();
  }
#pragma unroll
  for (int i = 0; i < 4; ++i)
#pragma unroll
    for (int j = 0; j < 2; ++j) {
      const int rb = brow + wm * 64 + i * 16 + ((lane & 16) ? 8 : 0);
      const int cc = bcol + wn * 32 + j * 16 + (lane & 15);
#pragma unroll
      for (int r = 0; r < 8; ++r)
        store_out(C + (size_t)(rb + r) * ldc + cc, acc[i][j][r]);
    }
}

// ---- K2: scores[bh] = q[bh] @ k[bh]^T / 8 + prev, causal -inf ------------
// q,k stored bf16 as [B][L][H*DK]; per (b,h) row stride = HD_.
// Tile staging uses CDNA5 async global->LDS DMA (verbatim bf16 copy).
__global__ __launch_bounds__(256)
void scores_kernel(const __bf16* __restrict__ q, const __bf16* __restrict__ k,
                   const float* __restrict__ prev, float* __restrict__ scores) {
  __shared__ __bf16 SMEM[2 * 128 * 72];     // [Ql | Kl], row stride 72 (16B aligned)
  __bf16* Ql = SMEM;
  __bf16* Kl = SMEM + 128 * 72;
  const int t = threadIdx.x, lane = t & 31, wid = t >> 5;
  const int wm = wid >> 2, wn = wid & 3;
  const int bh = blockIdx.z, b = bh >> 4, h = bh & 15;
  const int q0 = blockIdx.y * 128, kr0 = blockIdx.x * 128;
  const __bf16* qb = q + (size_t)b * L_ * HD_ + h * DK_;
  const __bf16* kb = k + (size_t)b * L_ * HD_ + h * DK_;
  {
    const int r0 = t >> 3, c = (t & 7) * 8;
#pragma unroll
    for (int i = 0; i < 4; ++i) {
      const int r = r0 + i * 32;
      async_load_b128(Ql + r * 72 + c, qb + (size_t)(q0 + r) * HD_ + c);
      async_load_b128(Kl + r * 72 + c, kb + (size_t)(kr0 + r) * HD_ + c);
    }
  }
  async_wait0();
  __syncthreads();
  v8f acc[4][2] = {};
#pragma unroll
  for (int kc = 0; kc < DK_; kc += 32) {
    v16bf af[4], bfr[2];
#pragma unroll
    for (int i = 0; i < 4; ++i) af[i] = frag_a(Ql + kc, wm * 64 + i * 16, lane, 72);
#pragma unroll
    for (int j = 0; j < 2; ++j) bfr[j] = frag_b(Kl + kc, wn * 32 + j * 16, lane, 72);
#pragma unroll
    for (int i = 0; i < 4; ++i)
#pragma unroll
      for (int j = 0; j < 2; ++j)
        acc[i][j] = wmma_bf16(af[i], bfr[j], acc[i][j]);
  }
  const float* pv = prev   + (size_t)bh * L_ * L_;
  float*       sc = scores + (size_t)bh * L_ * L_;
#pragma unroll
  for (int i = 0; i < 4; ++i)
#pragma unroll
    for (int j = 0; j < 2; ++j) {
      const int rb = q0  + wm * 64 + i * 16 + ((lane & 16) ? 8 : 0);
      const int cc = kr0 + wn * 32 + j * 16 + (lane & 15);
#pragma unroll
      for (int r = 0; r < 8; ++r) {
        const int row = rb + r;
        float v = acc[i][j][r] * 0.125f + pv[(size_t)row * L_ + cc];
        if (cc > row) v = -__builtin_inff();   // causal mask (triu, k=1)
        sc[(size_t)row * L_ + cc] = v;
      }
    }
}

// ---- K3: row-wise softmax (scores -> attn) -------------------------------
__global__ __launch_bounds__(256)
void softmax_kernel(const float* __restrict__ scores, float* __restrict__ attn) {
  const size_t rbase = (size_t)blockIdx.x * L_;
  const int t = threadIdx.x, lane = t & 31, wid = t >> 5;
  __shared__ float red[8];
  float x[4];
#pragma unroll
  for (int i = 0; i < 4; ++i) x[i] = scores[rbase + t + i * 256];
  float m = fmaxf(fmaxf(x[0], x[1]), fmaxf(x[2], x[3]));
#pragma unroll
  for (int o = 16; o > 0; o >>= 1) m = fmaxf(m, __shfl_xor(m, o, 32));
  if (lane == 0) red[wid] = m;
  __syncthreads();
  m = red[0];
#pragma unroll
  for (int i = 1; i < 8; ++i) m = fmaxf(m, red[i]);
  float s = 0.f;
#pragma unroll
  for (int i = 0; i < 4; ++i) { x[i] = expf(x[i] - m); s += x[i]; }
#pragma unroll
  for (int o = 16; o > 0; o >>= 1) s += __shfl_xor(s, o, 32);
  __syncthreads();
  if (lane == 0) red[wid] = s;
  __syncthreads();
  s = red[0];
#pragma unroll
  for (int i = 1; i < 8; ++i) s += red[i];
  const float inv = 1.0f / s;
#pragma unroll
  for (int i = 0; i < 4; ++i) attn[rbase + t + i * 256] = x[i] * inv;
}

// ---- K4: context[bh] = attn[bh] @ v[bh] ----------------------------------
// 128(M) x 64(N) block tile, 8 waves along M (16 rows each).
__global__ __launch_bounds__(256)
void context_kernel(const float* __restrict__ attn, const __bf16* __restrict__ v,
                    __bf16* __restrict__ ctx) {
  __shared__ __bf16 SMEM[128 * 40 + 64 * 40];
  __bf16* Al = SMEM;
  __bf16* Bl = SMEM + 128 * 40;
  const int t = threadIdx.x, lane = t & 31, wid = t >> 5;
  const int bh = blockIdx.y, b = bh >> 4, h = bh & 15;
  const int m0 = blockIdx.x * 128;
  const float*  ab = attn + (size_t)bh * L_ * L_ + (size_t)m0 * L_;
  const __bf16* vb = v + (size_t)b * L_ * HD_ + h * DV_;
  v8f acc[4] = {};
  for (int k0 = 0; k0 < L_; k0 += 32) {
    load_tile_a(ab + k0, Al, L_, t);               // attn fp32 -> bf16 LDS
    {                                              // v bf16, transpose -> [n][k]
      const int kk = t & 31, n0 = (t >> 5) * 8;
      v8bf val = *(const v8bf*)(vb + (size_t)(k0 + kk) * HD_ + n0);
#pragma unroll
      for (int j = 0; j < 8; ++j) Bl[(n0 + j) * 40 + kk] = val[j];
    }
    if (k0 + 32 < L_) {
      __builtin_prefetch(ab + (k0 + 32), 0, 0);
      __builtin_prefetch(vb + (size_t)(k0 + 32) * HD_, 0, 0);
    }
    __syncthreads();
    v16bf a = frag_a(Al, wid * 16, lane, 40);
#pragma unroll
    for (int j = 0; j < 4; ++j)
      acc[j] = wmma_bf16(a, frag_b(Bl, j * 16, lane, 40), acc[j]);
    __syncthreads();
  }
  const int rb = m0 + wid * 16 + ((lane & 16) ? 8 : 0);
#pragma unroll
  for (int j = 0; j < 4; ++j) {
    const int cc = j * 16 + (lane & 15);
#pragma unroll
    for (int r = 0; r < 8; ++r)
      ctx[((size_t)b * L_ + rb + r) * HD_ + h * DV_ + cc] = (__bf16)acc[j][r];
  }
}

// --------------------------------------------------------------------------
extern "C" void kernel_launch(void* const* d_in, const int* in_sizes, int n_in,
                              void* d_out, int out_size, void* d_ws, size_t ws_size,
                              hipStream_t stream) {
  (void)in_sizes; (void)n_in; (void)out_size; (void)ws_size;
  const float* Q    = (const float*)d_in[0];
  const float* K    = (const float*)d_in[1];
  const float* V    = (const float*)d_in[2];
  const float* prev = (const float*)d_in[3];
  // d_in[4] = attn_mask (causal triu) — reconstructed analytically, unused.
  const float* W_Q  = (const float*)d_in[5];
  const float* W_K  = (const float*)d_in[6];
  const float* W_V  = (const float*)d_in[7];
  const float* W_O  = (const float*)d_in[8];

  float* out    = (float*)d_out;                                 // [B,L,D]
  float* attn   = out  + (size_t)B_ * L_ * D_;                   // [B,H,L,L]
  float* scores = attn + (size_t)B_ * H_ * L_ * L_;              // [B,H,L,L]

  const size_t NP = (size_t)B_ * L_ * HD_;                       // 8M elems
  __bf16* q_bf = (__bf16*)d_ws;
  __bf16* k_bf = q_bf + NP;
  __bf16* v_bf = k_bf + NP;
  __bf16* c_bf = v_bf + NP;

  const dim3 blk(256);
  const dim3 gProj(D_ / 128, (B_ * L_) / 128);                   // (8, 64)

  // K1: QKV projections (fp32 -> bf16 out, fp32 accumulate in WMMA)
  gemm_kernel<float, __bf16><<<gProj, blk, 0, stream>>>(Q, W_Q, q_bf, D_, D_, HD_, HD_);
  gemm_kernel<float, __bf16><<<gProj, blk, 0, stream>>>(K, W_K, k_bf, D_, D_, HD_, HD_);
  gemm_kernel<float, __bf16><<<gProj, blk, 0, stream>>>(V, W_V, v_bf, D_, D_, HD_, HD_);

  // K2: scores = q k^T / sqrt(64) + prev, causal mask -> -inf (async-DMA tiles)
  scores_kernel<<<dim3(L_ / 128, L_ / 128, B_ * H_), blk, 0, stream>>>(q_bf, k_bf, prev, scores);

  // K3: attn = softmax(scores) rowwise
  softmax_kernel<<<dim3(B_ * H_ * L_), blk, 0, stream>>>(scores, attn);

  // K4: context = attn @ v  (attn read back fp32, converted to bf16 in LDS)
  context_kernel<<<dim3(L_ / 128, B_ * H_), blk, 0, stream>>>(attn, v_bf, c_bf);

  // K5: output = context @ W_O  (A-tiles staged via async global->LDS DMA)
  gemm_kernel<__bf16, float><<<gProj, blk, 0, stream>>>(c_bf, W_O, out, HD_, D_, D_, D_);
}
